// GATLocalizationModule_19172734009869
// MI455X (gfx1250) — matrix operand order repaired and checked by reading, not compiled, not run on hardware
//
#include <hip/hip_runtime.h>
#include <hip/hip_bf16.h>

// ---------------------------------------------------------------------------
// GAT localization module for MI455X (gfx1250, wave32).
// GEMMs: v_wmma_f32_16x16x32_bf16, double-buffered async global->LDS staging
// (global_load_async_to_lds_b128, s_wait_asynccnt 3 -> loads overlap WMMA).
// Edge softmax/aggregation: f32 atomics; h (17.9MB) stays L2-resident.
// ---------------------------------------------------------------------------

typedef __attribute__((ext_vector_type(16))) __bf16 v16bf;
typedef __attribute__((ext_vector_type(8)))  float  v8f;
typedef unsigned short ushort_t;

static constexpr int NN   = 20000;  // nodes
static constexpr int IND  = 128;    // input dim (32-multiple)
static constexpr int HID  = 200;    // hidden dim
static constexpr int LD   = 224;    // hidden padded to 32-multiple
static constexpr int F1   = 100, F1P = 128;
static constexpr int F2   = 50,  F2P = 64;

template<bool B> struct BoolC { static constexpr bool value = B; };

__device__ __forceinline__ ushort_t f2bf(float f) {
  unsigned u = __float_as_uint(f);
  u += 0x7FFFu + ((u >> 16) & 1u);          // round-to-nearest-even
  return (ushort_t)(u >> 16);
}
__device__ __forceinline__ float bf2f(ushort_t h) {
  return __uint_as_float(((unsigned)h) << 16);
}
__device__ __forceinline__ void atomicMaxF(float* addr, float val) {
  if (val >= 0.0f) atomicMax((int*)addr, __float_as_int(val));
  else             atomicMin((unsigned int*)addr, __float_as_uint(val));
}

// ---------------------------------------------------------------------------
// bf16 WMMA GEMM: C = act(A[M x Kpad] @ Bt^T + bias). A is zero-padded bf16,
// Bt is pre-transposed zero-padded bf16 [Npad64 x Kpad] -> staging is pure
// unconditional 16B chunks. Block tile 128x64, 256 threads (8 waves).
// Double-buffered LDS; next tile's async loads stay in flight during WMMA.
// ---------------------------------------------------------------------------
#define BM 128
#define BN 64
#define BK 32
#define KP 40   // LDS row pitch (ushorts): 80B, 16B-aligned, conflict-light

template<int ACT, int OUTBF>   // ACT: 0 none, 1 relu; OUTBF: 0 f32 out, 1 bf16 out
__global__ __launch_bounds__(256)
void gemm_bf16(const ushort_t* __restrict__ A, const ushort_t* __restrict__ Bt,
               const float* __restrict__ bias, void* __restrict__ Cout,
               int M, int lda, int ldbt, int Nact, int ldc, int nK)
{
  __shared__ __align__(16) ushort_t As[2][BM * KP];
  __shared__ __align__(16) ushort_t Bs[2][BN * KP];

  const int tid  = threadIdx.x;
  const int lane = tid & 31;
  const int wave = tid >> 5;
  const int mBase = blockIdx.x * BM;
  const int nBase = blockIdx.y * BN;
  const bool fullM = (mBase + BM) <= M;   // uniform per block

  v8f acc[4] = {v8f{}, v8f{}, v8f{}, v8f{}};

  const int n0 = lane & 15;
  const int hb = (lane & 16) ? 8 : 0;    // A K-half base (ISA 16-bit A layout)
  const int kb = (lane & 16) ? 8 : 0;    // B K-half base
  const int ar = (wave << 4) + (lane & 15);

  // per-thread staging coordinates (16B = 8 bf16 chunks)
  const int bR  = tid >> 2;              // B: n row 0..63
  const int kc  = (tid & 3) << 3;        // k chunk 0/8/16/24
  const int aR0 = tid >> 2;              // A rows 0..63
  const int aR1 = aR0 + 64;              // A rows 64..127

  // hoisted global pointers (advance by BK elements per K-step)
  const ushort_t* gB  = Bt + (size_t)(nBase + bR) * ldbt + kc;
  const ushort_t* gA0 = A + (size_t)(mBase + aR0) * lda + kc;
  const ushort_t* gA1 = A + (size_t)(mBase + aR1) * lda + kc;
  const bool vA0 = (mBase + aR0) < M;
  const bool vA1 = (mBase + aR1) < M;

  unsigned lB[2], lA0[2], lA1[2];
  #pragma unroll
  for (int b = 0; b < 2; ++b) {
    lB[b]  = (unsigned)(uintptr_t)&Bs[b][bR  * KP + kc];
    lA0[b] = (unsigned)(uintptr_t)&As[b][aR0 * KP + kc];
    lA1[b] = (unsigned)(uintptr_t)&As[b][aR1 * KP + kc];
  }

  auto run = [&](auto FULLC) {
    constexpr bool FULL = decltype(FULLC)::value;

    auto issue = [&](int kt) {
      const int b = kt & 1;
      const size_t ko = (size_t)kt * BK;
      asm volatile("global_load_async_to_lds_b128 %0, %1, off"
                   :: "v"(lB[b]), "v"(gB + ko) : "memory");
      if (FULL) {
        asm volatile("global_load_async_to_lds_b128 %0, %1, off"
                     :: "v"(lA0[b]), "v"(gA0 + ko) : "memory");
        asm volatile("global_load_async_to_lds_b128 %0, %1, off"
                     :: "v"(lA1[b]), "v"(gA1 + ko) : "memory");
      } else {
        uint4 z = {0u, 0u, 0u, 0u};
        uint4 v0 = z, v1 = z;
        if (vA0) v0 = *(const uint4*)(gA0 + ko);
        if (vA1) v1 = *(const uint4*)(gA1 + ko);
        *(uint4*)&As[b][aR0 * KP + kc] = v0;
        *(uint4*)&As[b][aR1 * KP + kc] = v1;
      }
    };

    issue(0);
    for (int kt = 0; kt < nK; ++kt) {
      const int b = kt & 1;
      __syncthreads();                       // everyone done reading buf b^1
      if (kt + 1 < nK) {
        issue(kt + 1);                       // prefetch into buf b^1
        if (FULL) asm volatile("s_wait_asynccnt 0x3" ::: "memory");
        else      asm volatile("s_wait_asynccnt 0x1" ::: "memory");
      } else {
        asm volatile("s_wait_asynccnt 0x0" ::: "memory");
      }
      __syncthreads();                       // buf b visible to all waves

      // ---- A fragment (ISA 16-bit A 16x32 layout) ----
      union { v16bf v; uint4 q[2]; } af;
      {
        const ushort_t* ap = &As[b][ar * KP + hb];
        af.q[0] = *(const uint4*)(ap);       // K = hb .. hb+7
        af.q[1] = *(const uint4*)(ap + 16);  // K = hb+16 .. hb+23
      }
      // ---- 4 B fragments + WMMA ----
      #pragma unroll
      for (int t = 0; t < 4; ++t) {
        union { v16bf v; uint4 q[2]; } bfu;
        const ushort_t* bp = &Bs[b][(t * 16 + n0) * KP + kb];
        bfu.q[0] = *(const uint4*)(bp);      // K = kb .. kb+7
        bfu.q[1] = *(const uint4*)(bp + 16); // K = kb+16 .. kb+23
        acc[t] = __builtin_amdgcn_wmma_f32_16x16x32_bf16(
                     false, af.v, false, bfu.v, (short)0, acc[t], false, false);
      }
    }
  };
  if (fullM) run(BoolC<true>{}); else run(BoolC<false>{});

  // ---- epilogue: bias + activation; padded cols get exact 0 ----
  const int rb = mBase + (wave << 4) + ((lane & 16) ? 8 : 0);
  #pragma unroll
  for (int t = 0; t < 4; ++t) {
    int col = nBase + t * 16 + (lane & 15);
    if (col >= ldc) continue;
    float bb = (bias != nullptr && col < Nact) ? bias[col] : 0.0f;
    #pragma unroll
    for (int v = 0; v < 8; ++v) {
      int row = rb + v;
      if (row < M) {
        float val = acc[t][v] + bb;
        if (ACT == 1) val = fmaxf(val, 0.0f);
        if (OUTBF) ((ushort_t*)Cout)[(size_t)row * ldc + col] = f2bf(val);
        else       ((float*)Cout)[(size_t)row * ldc + col] = val;
      }
    }
  }
}

// ---------------------------------------------------------------------------
// conversion kernels (run once per launch; tiny)
// ---------------------------------------------------------------------------
__global__ void convert_bf_k(const float* __restrict__ in, ushort_t* __restrict__ out, int n) {
  int i = blockIdx.x * 256 + threadIdx.x;
  if (i < n) out[i] = f2bf(in[i]);
}

// W[K x N] (row-major) -> Wt[Nrows x Kp] bf16, transposed + zero padded
__global__ void convert_wt_k(const float* __restrict__ W, ushort_t* __restrict__ out,
                             int K, int N, int Kp, int Nrows) {
  int i = blockIdx.x * 256 + threadIdx.x;
  int tot = Nrows * Kp;
  if (i >= tot) return;
  int n = i / Kp, k = i - n * Kp;
  out[i] = (n < N && k < K) ? f2bf(W[(size_t)k * N + n]) : (ushort_t)0;
}

// ---------------------------------------------------------------------------
// GAT edge-phase kernels
// ---------------------------------------------------------------------------
__global__ void fill_k(float* __restrict__ p, size_t n, float v) {
  size_t i = (size_t)blockIdx.x * 256 + threadIdx.x;
  if (i < n) p[i] = v;
}

// wave-per-node: a_src[i] = h[i].as ; a_dst[i] = h[i].ad
__global__ __launch_bounds__(256)
void attn_dots_k(const float* __restrict__ h, const float* __restrict__ as,
                 const float* __restrict__ ad, float* __restrict__ asrc,
                 float* __restrict__ adst, int n) {
  int lane = threadIdx.x & 31;
  int node = blockIdx.x * 8 + (threadIdx.x >> 5);
  if (node >= n) return;
  const float* hr = h + (size_t)node * LD;
  float s0 = 0.0f, s1 = 0.0f;
  for (int c = lane; c < HID; c += 32) {
    float v = hr[c];
    s0 += v * as[c];
    s1 += v * ad[c];
  }
  for (int off = 16; off; off >>= 1) {
    s0 += __shfl_xor(s0, off, 32);
    s1 += __shfl_xor(s1, off, 32);
  }
  if (lane == 0) { asrc[node] = s0; adst[node] = s1; }
}

// pass 1: e = leaky_relu(a_src[s] + a_dst[d], 0.2); m[d] = max(m[d], e)
__global__ void edge_max_k(const float* __restrict__ asrc, const float* __restrict__ adst,
                           const int* __restrict__ src, const int* __restrict__ dst,
                           int E, int ET, float* __restrict__ ebuf, float* __restrict__ mbuf) {
  int i = blockIdx.x * 256 + threadIdx.x;
  if (i >= ET) return;
  int s, d;
  if (i < E) { s = src[i]; d = dst[i]; } else { s = d = i - E; }
  float e = asrc[s] + adst[d];
  e = e > 0.0f ? e : 0.2f * e;
  ebuf[i] = e;
  atomicMaxF(&mbuf[d], e);
}

// pass 2: w = exp(e - m[d]); s[d] += w
__global__ void edge_exp_k(const float* __restrict__ ebuf, const int* __restrict__ dst,
                           int E, int ET, const float* __restrict__ mbuf,
                           float* __restrict__ wbuf, float* __restrict__ sbuf) {
  int i = blockIdx.x * 256 + threadIdx.x;
  if (i >= ET) return;
  int d = (i < E) ? dst[i] : i - E;
  float w = __expf(ebuf[i] - mbuf[d]);
  wbuf[i] = w;
  atomicAdd(&sbuf[d], w);
}

// wave-per-edge: out[d] += (w/s[d]) * h[s]   (h rows L2-resident)
__global__ __launch_bounds__(256)
void aggregate_k(const float* __restrict__ h, const float* __restrict__ wbuf,
                 const float* __restrict__ sbuf, const int* __restrict__ src,
                 const int* __restrict__ dst, int E, int ET, float* __restrict__ out) {
  int lane = threadIdx.x & 31;
  int e = blockIdx.x * 8 + (threadIdx.x >> 5);
  if (e >= ET) return;
  int s, d;
  if (e < E) { s = src[e]; d = dst[e]; } else { s = d = e - E; }
  const float* hr = h + (size_t)s * LD;
  __builtin_prefetch(hr + lane, 0, 3);            // global_prefetch_b8
  __builtin_prefetch(hr + 128 + lane, 0, 3);
  float alpha = wbuf[e] / sbuf[d];
  float* orow = out + (size_t)d * LD;
  for (int c = lane; c < HID; c += 32)
    atomicAdd(&orow[c], alpha * hr[c]);
}

// one block per node row: y_bf16 = leaky_relu(x + bias, 0.01); pad cols -> 0
__global__ __launch_bounds__(LD)
void bias_leaky_bf_k(const float* __restrict__ in, const float* __restrict__ bias,
                     ushort_t* __restrict__ out) {
  int col = threadIdx.x;
  size_t i = (size_t)blockIdx.x * LD + col;
  float v = in[i] + (col < HID ? bias[col] : 0.0f);
  v = v > 0.0f ? v : 0.01f * v;
  out[i] = f2bf(v);
}

// final 50 -> 2 head: thread per node, bf16 input
__global__ void fc3_k(const ushort_t* __restrict__ h, const float* __restrict__ w,
                      const float* __restrict__ b, float* __restrict__ out, int n) {
  int node = blockIdx.x * 256 + threadIdx.x;
  if (node >= n) return;
  const ushort_t* hr = h + (size_t)node * F2P;
  float a0 = b[0], a1 = b[1];
  #pragma unroll 5
  for (int k = 0; k < F2; ++k) {
    float v = bf2f(hr[k]);
    a0 += v * w[2 * k];
    a1 += v * w[2 * k + 1];
  }
  out[2 * node]     = a0;
  out[2 * node + 1] = a1;
}

// ---------------------------------------------------------------------------
extern "C" void kernel_launch(void* const* d_in, const int* in_sizes, int n_in,
                              void* d_out, int out_size, void* d_ws, size_t ws_size,
                              hipStream_t stream) {
  const float* x    = (const float*)d_in[0];
  const int*   ei   = (const int*)d_in[1];
  const float* W1   = (const float*)d_in[2];
  const float* as1  = (const float*)d_in[3];
  const float* ad1  = (const float*)d_in[4];
  const float* b1   = (const float*)d_in[5];
  const float* W2   = (const float*)d_in[6];
  const float* as2  = (const float*)d_in[7];
  const float* ad2  = (const float*)d_in[8];
  const float* b2   = (const float*)d_in[9];
  const float* fc1w = (const float*)d_in[10];
  const float* fc1b = (const float*)d_in[11];
  const float* fc2w = (const float*)d_in[12];
  const float* fc2b = (const float*)d_in[13];
  const float* fc3w = (const float*)d_in[14];
  const float* fc3b = (const float*)d_in[15];
  float* out = (float*)d_out;

  const int E  = in_sizes[1] / 2;     // 320000 raw edges
  const int ET = E + NN;              // + self loops
  const int* src = ei;
  const int* dst = ei + E;

  // ---- workspace layout ----
  char* base = (char*)d_ws;
  size_t off = 0;
  auto carve = [&](size_t bytes) { void* p = base + off; off = (off + bytes + 255) & ~(size_t)255; return p; };

  const size_t NL = (size_t)NN * LD;
  float*    X    = (float*)   carve(NL * 4);              // h_lin (f32)
  float*    Z    = (float*)   carve(NL * 4);              // aggregation target
  float*    asrc = (float*)   carve((size_t)NN * 4);
  float*    adst = (float*)   carve((size_t)NN * 4);
  float*    mbuf = (float*)   carve((size_t)NN * 4);
  float*    sbuf = (float*)   carve((size_t)NN * 4);
  float*    ebuf = (float*)   carve((size_t)ET * 4);
  float*    wbuf = (float*)   carve((size_t)ET * 4);
  ushort_t* xb   = (ushort_t*)carve((size_t)NN * IND * 2);
  ushort_t* Yb   = (ushort_t*)carve(NL * 2);               // bf16 activations
  ushort_t* h3b  = (ushort_t*)carve((size_t)NN * F1P * 2);
  ushort_t* h4b  = (ushort_t*)carve((size_t)NN * F2P * 2);
  ushort_t* W1t  = (ushort_t*)carve((size_t)256 * IND * 2);
  ushort_t* W2t  = (ushort_t*)carve((size_t)256 * LD  * 2);
  ushort_t* f1t  = (ushort_t*)carve((size_t)F1P * LD  * 2);
  ushort_t* f2t  = (ushort_t*)carve((size_t)F2P * F1P * 2);

  const dim3 blk(256);
  const dim3 gHn((NN + BM - 1) / BM, (LD  + BN - 1) / BN);  // 157 x 4
  const dim3 gF1((NN + BM - 1) / BM, (F1P + BN - 1) / BN);  // 157 x 2
  const dim3 gF2((NN + BM - 1) / BM, (F2P + BN - 1) / BN);  // 157 x 1
  const int gNode  = (NN + 7) / 8;
  const int gEdgeT = (ET + 255) / 256;
  const int gEdgeW = (ET + 7) / 8;
  const float NEG_INF = -__builtin_huge_valf();

  // ---- one-time conversions (bf16, transpose+pad weights) ----
  convert_bf_k<<<(NN * IND + 255) / 256, blk, 0, stream>>>(x, xb, NN * IND);
  convert_wt_k<<<(256 * IND + 255) / 256, blk, 0, stream>>>(W1,   W1t, IND, HID, IND, 256);
  convert_wt_k<<<(256 * LD  + 255) / 256, blk, 0, stream>>>(W2,   W2t, HID, HID, LD,  256);
  convert_wt_k<<<(F1P * LD  + 255) / 256, blk, 0, stream>>>(fc1w, f1t, HID, F1,  LD,  F1P);
  convert_wt_k<<<(F2P * F1P + 255) / 256, blk, 0, stream>>>(fc2w, f2t, F1,  F2,  F1P, F2P);

  // ---------- GAT layer 1 ----------
  gemm_bf16<0, 0><<<gHn, blk, 0, stream>>>(xb, W1t, nullptr, X, NN, IND, IND, HID, LD, IND / BK);
  fill_k<<<(NN + 255) / 256, blk, 0, stream>>>(mbuf, NN, NEG_INF);
  fill_k<<<(NN + 255) / 256, blk, 0, stream>>>(sbuf, NN, 0.0f);
  fill_k<<<(int)((NL + 255) / 256), blk, 0, stream>>>(Z, NL, 0.0f);
  attn_dots_k<<<gNode, blk, 0, stream>>>(X, as1, ad1, asrc, adst, NN);
  edge_max_k<<<gEdgeT, blk, 0, stream>>>(asrc, adst, src, dst, E, ET, ebuf, mbuf);
  edge_exp_k<<<gEdgeT, blk, 0, stream>>>(ebuf, dst, E, ET, mbuf, wbuf, sbuf);
  aggregate_k<<<gEdgeW, blk, 0, stream>>>(X, wbuf, sbuf, src, dst, E, ET, Z);
  bias_leaky_bf_k<<<NN, dim3(LD), 0, stream>>>(Z, b1, Yb);

  // ---------- GAT layer 2 ----------
  gemm_bf16<0, 0><<<gHn, blk, 0, stream>>>(Yb, W2t, nullptr, X, NN, LD, LD, HID, LD, LD / BK);
  fill_k<<<(NN + 255) / 256, blk, 0, stream>>>(mbuf, NN, NEG_INF);
  fill_k<<<(NN + 255) / 256, blk, 0, stream>>>(sbuf, NN, 0.0f);
  fill_k<<<(int)((NL + 255) / 256), blk, 0, stream>>>(Z, NL, 0.0f);
  attn_dots_k<<<gNode, blk, 0, stream>>>(X, as2, ad2, asrc, adst, NN);
  edge_max_k<<<gEdgeT, blk, 0, stream>>>(asrc, adst, src, dst, E, ET, ebuf, mbuf);
  edge_exp_k<<<gEdgeT, blk, 0, stream>>>(ebuf, dst, E, ET, mbuf, wbuf, sbuf);
  aggregate_k<<<gEdgeW, blk, 0, stream>>>(X, wbuf, sbuf, src, dst, E, ET, Z);
  bias_leaky_bf_k<<<NN, dim3(LD), 0, stream>>>(Z, b2, Yb);

  // ---------- FC head (bf16 in / bf16 out, fused bias+relu) ----------
  gemm_bf16<1, 1><<<gF1, blk, 0, stream>>>(Yb,  f1t, fc1b, h3b, NN, LD,  LD,  F1, F1P, LD / BK);
  gemm_bf16<1, 1><<<gF2, blk, 0, stream>>>(h3b, f2t, fc2b, h4b, NN, F1P, F1P, F2, F2P, F1P / BK);
  fc3_k<<<(NN + 255) / 256, blk, 0, stream>>>(h4b, fc3w, fc3b, out, NN);
}
